// BiLSTMTagger_55662776156722
// MI455X (gfx1250) — compile-verified, matrix operand order checked by reading, not run on hardware
//
#include <hip/hip_runtime.h>
#include <hip/hip_bf16.h>

// BiLSTM tagger for MI455X (gfx1250).
//  - bf16 WMMA (v_wmma_f32_16x16x32_bf16) for the two parallel GEMMs
//  - fp32 VALU GEMV + CDNA5 cluster split-barrier for the sequential scan,
//    splitting each step's [1,512]x[512,2048] across a 16-WGP cluster.
//  - Each scan WG parks its 128x512 fp32 W_hh slice in LDS (~266 KB of the
//    320 KB/WGP) via the Tensor Data Mover, using TDM's pad feature
//    (pad_interval=256 dw, pad_amount=4 dw) to kill LDS bank conflicts.
//    (Dispatch should set cluster dims (16,1,1); compile-only here.)

typedef __bf16 bf16;
typedef __attribute__((ext_vector_type(16))) __bf16 bf16x16;
typedef __attribute__((ext_vector_type(8)))  __bf16 bf16x8;
typedef __attribute__((ext_vector_type(8)))  float  f32x8;
typedef __attribute__((ext_vector_type(4)))  unsigned int v4u;
typedef __attribute__((ext_vector_type(8)))  int v8i_t;
typedef __attribute__((ext_vector_type(4)))  int v4i_t;

#define TT   4096
#define EE   256
#define HH   512
#define G4   2048   // 4*H
#define NTAGS 50
#define KOUT 1024   // 2*H

// LDS row pitch for the W_hh slice: 512 data floats + 2x4 pad floats
#define WPITCH 520
#define WHALF  260

// ---------------- workspace layout (bytes) ----------------
constexpr size_t OFF_EMB  = 0;
constexpr size_t SZ_EMB   = (size_t)TT * EE * 2;            // embeds bf16 [T,E]
constexpr size_t OFF_WIHF = OFF_EMB + SZ_EMB;
constexpr size_t SZ_WIH   = (size_t)G4 * EE * 2;            // W_ih bf16 [4H,E]
constexpr size_t OFF_WIHB = OFF_WIHF + SZ_WIH;
constexpr size_t OFF_WOUT = OFF_WIHB + SZ_WIH;
constexpr size_t SZ_WOUT  = (size_t)64 * KOUT * 2;          // W_out bf16 padded [64,1024]
constexpr size_t OFF_BSF  = OFF_WOUT + SZ_WOUT;
constexpr size_t SZ_BS    = (size_t)G4 * 4;                 // b_ih+b_hh fp32 [4H]
constexpr size_t OFF_BSB  = OFF_BSF + SZ_BS;
constexpr size_t OFF_GINF = OFF_BSB + SZ_BS;
constexpr size_t SZ_GIN   = (size_t)TT * G4 * 4;            // gate inputs fp32 [T,4H]
constexpr size_t OFF_GINB = OFF_GINF + SZ_GIN;
constexpr size_t OFF_HF   = OFF_GINB + SZ_GIN;
constexpr size_t SZ_HBUF  = (size_t)(TT + 1) * HH * 4;      // h states fp32 [T+1,H]
constexpr size_t OFF_HB   = OFF_HF + SZ_HBUF;
constexpr size_t OFF_OUTS = OFF_HB + SZ_HBUF;               // concat bf16 [T,2H]

// ---------------- prep kernels ----------------
__global__ void k_gather_embed(const int* __restrict__ sent,
                               const float* __restrict__ emb,
                               bf16* __restrict__ out) {
    const int t = blockIdx.x;
    const int e = threadIdx.x;                 // 256 threads = 8 waves
    const int v = sent[t];
    out[(size_t)t * EE + e] = (bf16)emb[(size_t)v * EE + e];
}

__global__ void k_f32_to_bf16(const float* __restrict__ src, bf16* __restrict__ dst, int n) {
    const int i = blockIdx.x * blockDim.x + threadIdx.x;
    if (i < n) dst[i] = (bf16)src[i];
}

__global__ void k_pack_wout(const float* __restrict__ w, bf16* __restrict__ dst) {
    const int i = blockIdx.x * blockDim.x + threadIdx.x;   // 64*1024
    const int row = i >> 10, col = i & 1023;
    dst[i] = (row < NTAGS) ? (bf16)w[(size_t)row * KOUT + col] : (bf16)0.f;
}

__global__ void k_bias_sum(const float* __restrict__ a_f, const float* __restrict__ b_f,
                           const float* __restrict__ a_b, const float* __restrict__ b_b,
                           float* __restrict__ o_f, float* __restrict__ o_b) {
    const int i = blockIdx.x * blockDim.x + threadIdx.x;   // 2048
    o_f[i] = a_f[i] + b_f[i];
    o_b[i] = a_b[i] + b_b[i];
}

__global__ void k_init_h(float* __restrict__ hf, float* __restrict__ hb) {
    const int i = blockIdx.x * blockDim.x + threadIdx.x;   // 512
    hf[i] = 0.f;                       // h_f(-1) = 0 (row 0)
    hb[(size_t)TT * HH + i] = 0.f;     // h_b(T)  = 0 (row T)
}

// ---------------- WMMA fragment load ----------------
// ISA 16-bit A/B layout: lane&15 selects the M/N row, lanes>=16 hold the
// second K half; each lane holds K = {kb..kb+7, kb+16..kb+23} (two 16B loads).
union FragBF { bf16x16 v; bf16x8 h[2]; };

__device__ inline bf16x16 load_frag(const bf16* __restrict__ base, int row_stride,
                                    int lane, int kbase) {
    const bf16* p = base + (size_t)(lane & 15) * row_stride + kbase + ((lane >> 4) << 3);
    FragBF f;
    f.h[0] = *(const bf16x8*)(p);
    f.h[1] = *(const bf16x8*)(p + 16);
    return f.v;
}

// ---------------- gate-input GEMM: Gin = X @ W_ih^T + (b_ih+b_hh) ----------
// grid (4H/64, T/16, 2 dirs), block 128 (4 waves; one 16x16 tile each).
__global__ void k_gin_gemm(const bf16* __restrict__ x,
                           const bf16* __restrict__ wih_f, const bf16* __restrict__ wih_b,
                           const float* __restrict__ bs_f, const float* __restrict__ bs_b,
                           float* __restrict__ gin_f, float* __restrict__ gin_b) {
    const int lane = threadIdx.x & 31;
    const int wave = threadIdx.x >> 5;
    const int j0 = blockIdx.x * 64 + wave * 16;
    const int t0 = blockIdx.y * 16;
    const bf16*  w   = blockIdx.z ? wih_b : wih_f;
    const float* bs  = blockIdx.z ? bs_b  : bs_f;
    float*       gin = blockIdx.z ? gin_b : gin_f;

    const bf16* xrow = x + (size_t)t0 * EE;
    const bf16* wrow = w + (size_t)j0 * EE;
    f32x8 acc = {};
#pragma unroll
    for (int k = 0; k < EE; k += 32) {
        bf16x16 a = load_frag(xrow, EE, lane, k);
        bf16x16 b = load_frag(wrow, EE, lane, k);
        acc = __builtin_amdgcn_wmma_f32_16x16x32_bf16(false, a, false, b,
                                                      (short)0, acc, false, false);
    }
    // C layout: col = lane&15, row = r + 8*(lane>=16)
    const int   col   = j0 + (lane & 15);
    const float bias  = bs[col];
    const int   rbase = t0 + ((lane >> 4) << 3);
#pragma unroll
    for (int r = 0; r < 8; ++r)
        gin[(size_t)(rbase + r) * G4 + col] = acc[r] + bias;
}

// ---------------- TDM: 2D tile of fp32 rows -> LDS with anti-conflict pad --
// D# per CDNA5 ISA: count=1, type=2 ("image"),
// data_size=4B, tile 512x32, tensor_dim0_stride=512,
// pad_enable=1, pad_interval=7 (256 dw), pad_amount=3 (4 dw)
// -> LDS row layout [256|4|256|4] floats (pitch 520, halves at +0/+260).
__device__ inline void tdm_load_whh_block(const float* __restrict__ gsrc,
                                          unsigned lds_byte_addr) {
#if __has_builtin(__builtin_amdgcn_tensor_load_to_lds)
    const unsigned long long ga = (unsigned long long)(uintptr_t)gsrc;
    v4u g0;
    g0[0] = 1u;                                               // count=1 (user D#)
    g0[1] = lds_byte_addr;                                    // lds_addr
    g0[2] = (unsigned)(ga & 0xFFFFFFFFu);                     // global_addr[31:0]
    g0[3] = (unsigned)((ga >> 32) & 0x01FFFFFFu) | (2u << 30);// addr[56:32] | type=2
    v8i_t g1;
    g1[0] = (int)((2u << 16) | (1u << 20) | (7u << 22) | (3u << 25));
           // data_size=4B | pad_enable | pad_interval=256dw | pad_amount=4dw
    g1[1] = (int)((512u & 0xFFFFu) << 16);                    // tensor_dim0 lo
    g1[2] = (int)(((512u >> 16) & 0xFFFFu) | (32u << 16));    // dim0 hi | tensor_dim1 lo
    g1[3] = (int)(0u | (512u << 16));                         // dim1 hi | tile_dim0=512
    g1[4] = (int)(32u);                                       // tile_dim1=32, tile_dim2=0
    g1[5] = (int)512;                                         // tensor_dim0_stride lo
    g1[6] = 0;                                                // stride hi | dim1_stride lo
    g1[7] = 0;
    v4i_t z4 = {0, 0, 0, 0};
#if defined(__clang_major__) && (__clang_major__ >= 23)
    v8i_t z8 = {0, 0, 0, 0, 0, 0, 0, 0};
    __builtin_amdgcn_tensor_load_to_lds(g0, g1, z4, z4, z8, 0);
#else
    __builtin_amdgcn_tensor_load_to_lds(g0, g1, z4, z4, 0);
#endif
#else
    (void)gsrc; (void)lds_byte_addr;   // fallback handled by caller
#endif
}

// ---------------- cluster-cooperative LSTM scan ----------------
// grid (16 WGs = one cluster, 2 dirs), block 256 (8 waves).
// WG `wg` owns h rows [32*wg, 32*wg+32) -> 128 gate rows (i,f,g,o blocks).
// Startup: TDM-copy the WG's 128x512 fp32 W_hh slice into LDS (4 descriptors).
// Per step: stage h_prev (2KB) in LDS, fp32 float4 GEMV out of LDS
// (256 MAC/thread), gate nonlinearity on 32 threads, publish h chunk to L2,
// cluster split-barrier.
__global__ void k_lstm_scan(const float* __restrict__ whh_f, const float* __restrict__ whh_b,
                            const float* __restrict__ gin_f, const float* __restrict__ gin_b,
                            float* __restrict__ hf, float* __restrict__ hb) {
    extern __shared__ float smem[];
    float* w_lds    = smem;                    // 128 * WPITCH floats (~266 KB)
    float* lds_h    = smem + 128 * WPITCH;     // 512
    float* lds_gate = lds_h + HH;              // 128
    float* lds_c    = lds_gate + 128;          // 32

    const int tid = threadIdx.x;
    const int wg  = blockIdx.x;            // workgroup-in-cluster
    const int dir = blockIdx.y;

    const float* whh  = dir ? whh_b : whh_f;
    const float* gin  = dir ? gin_b : gin_f;
    float*       hbuf = dir ? hb    : hf;

    // ---- stage W_hh slice into LDS via the Tensor Data Mover ----
#if __has_builtin(__builtin_amdgcn_tensor_load_to_lds)
    if (tid < 32) {                        // wave 0 issues 4 block descriptors
        const unsigned lds_base = (unsigned)(uintptr_t)(void*)w_lds;
#pragma unroll
        for (int g = 0; g < 4; ++g) {
            const float* gsrc = whh + (size_t)(g * HH + wg * 32) * HH;
            tdm_load_whh_block(gsrc, lds_base + (unsigned)g * 32u * WPITCH * 4u);
        }
        __builtin_amdgcn_s_wait_tensorcnt(0);
    }
#else
    for (int i = tid; i < 128 * HH; i += blockDim.x) {    // cooperative fallback
        const int row = i >> 9, col = i & 511;
        w_lds[row * WPITCH + (col >> 8) * WHALF + (col & 255)] =
            whh[(size_t)((row >> 5) * HH + wg * 32 + (row & 31)) * HH + col];
    }
#endif
    if (tid < 32) lds_c[tid] = 0.f;
    __syncthreads();

    const int r    = tid >> 1;             // local gate row 0..127
    const int half = tid & 1;              // which 256-wide half of the dot
    const int gidx = r >> 5;               // 0:i 1:f 2:g 3:o
    const int jloc = r & 31;
    const int grow = gidx * HH + wg * 32 + jloc;   // global gate row

    const float4* w4 = (const float4*)(w_lds + (size_t)r * WPITCH + half * WHALF);

    for (int s = 0; s < TT; ++s) {
        const int t = dir ? (TT - 1 - s) : s;
        // forward reads h(t-1) at row t; backward reads h(t+1) at row t+1
        const float* hp = hbuf + (size_t)(dir ? t + 1 : t) * HH;

        __syncthreads();                   // prior step done with lds_h
        lds_h[tid]       = hp[tid];
        lds_h[tid + 256] = hp[tid + 256];
        __syncthreads();

        // prefetch next step's gate-input row into cache hierarchy
        const int tn = dir ? (t > 0 ? t - 1 : t) : (t + 1 < TT ? t + 1 : t);
        __builtin_prefetch(gin + (size_t)tn * G4 + grow, 0, 1);

        const float4* h4 = (const float4*)(lds_h + half * 256);
        float ssum = 0.f;
#pragma unroll 8
        for (int k = 0; k < 64; ++k) {
            float4 wv = w4[k], hv = h4[k];
            ssum = fmaf(wv.x, hv.x, ssum);
            ssum = fmaf(wv.y, hv.y, ssum);
            ssum = fmaf(wv.z, hv.z, ssum);
            ssum = fmaf(wv.w, hv.w, ssum);
        }
        ssum += __shfl_xor(ssum, 1);       // combine halves (wave32)
        if (half == 0)
            lds_gate[r] = ssum + gin[(size_t)t * G4 + grow];
        __syncthreads();

        if (tid < 32) {
            const float gi = lds_gate[tid];
            const float gf = lds_gate[32 + tid];
            const float gg = lds_gate[64 + tid];
            const float go = lds_gate[96 + tid];
            const float ii = 1.f / (1.f + __expf(-gi));
            const float ff = 1.f / (1.f + __expf(-gf));
            const float oo = 1.f / (1.f + __expf(-go));
            const float c  = ff * lds_c[tid] + ii * tanhf(gg);
            lds_c[tid] = c;
            const float hn = oo * tanhf(c);
            // forward writes h(t) at row t+1; backward writes h(t) at row t
            hbuf[(size_t)(dir ? t : t + 1) * HH + wg * 32 + tid] = hn;
        }
        __threadfence();                   // release h chunk to device scope
        __syncthreads();
        // cluster split-barrier: one wave signals, all waves wait
        if (tid < 32) asm volatile("s_barrier_signal -3" ::: "memory");
        asm volatile("s_barrier_wait -3" ::: "memory");
    }
}

// ---------------- pack [h_f | h_b] -> bf16 [T, 2H] ----------------
__global__ void k_pack_outs(const float* __restrict__ hf, const float* __restrict__ hb,
                            bf16* __restrict__ outs) {
    const int t = blockIdx.x;
    for (int k = threadIdx.x; k < KOUT; k += blockDim.x) {
        const float v = (k < HH) ? hf[(size_t)(t + 1) * HH + k]
                                 : hb[(size_t)t * HH + (k - HH)];
        outs[(size_t)t * KOUT + k] = (bf16)v;
    }
}

// ---------------- tag projection: Y = outs @ W_out^T + b_out ----------------
// grid (T/16), block 128 (4 waves cover N=64 padded cols), K=1024.
__global__ void k_out_gemm(const bf16* __restrict__ outs, const bf16* __restrict__ wout,
                           const float* __restrict__ bout, float* __restrict__ y) {
    const int lane = threadIdx.x & 31;
    const int wave = threadIdx.x >> 5;
    const int n0 = wave * 16;
    const int t0 = blockIdx.x * 16;

    const bf16* arow = outs + (size_t)t0 * KOUT;
    const bf16* brow = wout + (size_t)n0 * KOUT;
    f32x8 acc = {};
#pragma unroll 4
    for (int k = 0; k < KOUT; k += 32) {
        bf16x16 a = load_frag(arow, KOUT, lane, k);
        bf16x16 b = load_frag(brow, KOUT, lane, k);
        acc = __builtin_amdgcn_wmma_f32_16x16x32_bf16(false, a, false, b,
                                                      (short)0, acc, false, false);
    }
    const int col = n0 + (lane & 15);
    if (col < NTAGS) {
        const float bias  = bout[col];
        const int   rbase = t0 + ((lane >> 4) << 3);
#pragma unroll
        for (int r = 0; r < 8; ++r)
            y[(size_t)(rbase + r) * NTAGS + col] = acc[r] + bias;
    }
}

// ---------------- host launcher ----------------
extern "C" void kernel_launch(void* const* d_in, const int* in_sizes, int n_in,
                              void* d_out, int out_size, void* d_ws, size_t ws_size,
                              hipStream_t stream) {
    (void)in_sizes; (void)n_in; (void)out_size; (void)ws_size;

    const int*   sentence = (const int*)  d_in[0];
    const float* emb      = (const float*)d_in[1];
    const float* W_ih_f   = (const float*)d_in[2];
    const float* W_hh_f   = (const float*)d_in[3];
    const float* b_ih_f   = (const float*)d_in[4];
    const float* b_hh_f   = (const float*)d_in[5];
    const float* W_ih_b   = (const float*)d_in[6];
    const float* W_hh_b   = (const float*)d_in[7];
    const float* b_ih_b   = (const float*)d_in[8];
    const float* b_hh_b   = (const float*)d_in[9];
    const float* W_out    = (const float*)d_in[10];
    const float* b_out    = (const float*)d_in[11];
    float* out = (float*)d_out;

    char* ws = (char*)d_ws;
    bf16*  embeds = (bf16*) (ws + OFF_EMB);
    bf16*  wihf   = (bf16*) (ws + OFF_WIHF);
    bf16*  wihb   = (bf16*) (ws + OFF_WIHB);
    bf16*  woutp  = (bf16*) (ws + OFF_WOUT);
    float* bsf    = (float*)(ws + OFF_BSF);
    float* bsb    = (float*)(ws + OFF_BSB);
    float* ginf   = (float*)(ws + OFF_GINF);
    float* ginb   = (float*)(ws + OFF_GINB);
    float* hfbuf  = (float*)(ws + OFF_HF);
    float* hbbuf  = (float*)(ws + OFF_HB);
    bf16*  outsb  = (bf16*) (ws + OFF_OUTS);

    // phase 0: gather/convert/fold
    k_gather_embed<<<TT, EE, 0, stream>>>(sentence, emb, embeds);
    k_f32_to_bf16<<<(G4 * EE) / 256, 256, 0, stream>>>(W_ih_f, wihf, G4 * EE);
    k_f32_to_bf16<<<(G4 * EE) / 256, 256, 0, stream>>>(W_ih_b, wihb, G4 * EE);
    k_pack_wout<<<(64 * KOUT) / 256, 256, 0, stream>>>(W_out, woutp);
    k_bias_sum<<<G4 / 256, 256, 0, stream>>>(b_ih_f, b_hh_f, b_ih_b, b_hh_b, bsf, bsb);
    k_init_h<<<HH / 256, 256, 0, stream>>>(hfbuf, hbbuf);

    // phase 1: WMMA gate-input GEMM, both directions
    k_gin_gemm<<<dim3(G4 / 64, TT / 16, 2), 128, 0, stream>>>(
        embeds, wihf, wihb, bsf, bsb, ginf, ginb);

    // phase 2: cluster-cooperative sequential scan (cluster dims (16,1,1)).
    // Dynamic LDS: W slice (128*WPITCH) + h (512) + gates (128) + c (32).
    const size_t smem_bytes = (size_t)(128 * WPITCH + HH + 128 + 32) * sizeof(float);
    k_lstm_scan<<<dim3(16, 2), 256, smem_bytes, stream>>>(
        W_hh_f, W_hh_b, ginf, ginb, hfbuf, hbbuf);

    // phase 3: pack + WMMA tag projection
    k_pack_outs<<<TT, 256, 0, stream>>>(hfbuf, hbbuf, outsb);
    k_out_gemm<<<TT / 16, 128, 0, stream>>>(outsb, woutp, b_out, out);
}
